// SNN_27960237097151
// MI455X (gfx1250) — matrix operand order repaired and checked by reading, not compiled, not run on hardware
//
#include <hip/hip_runtime.h>

typedef __attribute__((ext_vector_type(16))) _Float16 v16h;
typedef __attribute__((ext_vector_type(8)))  _Float16 v8h;
typedef __attribute__((ext_vector_type(8)))  float    v8f;
typedef __attribute__((ext_vector_type(4)))  unsigned int u32x4;
typedef __attribute__((ext_vector_type(8)))  int      i32x8;
typedef __attribute__((ext_vector_type(4)))  int      i32x4;

#define BETA      0.9f
#define THRESH    1.0f
#define NUM_STEPS 35
#define BSZ       16384
#define FD        1024
#define H1D       512
#define H2D       256
#define NCD       2

// ---- LDS layout (units: _Float16 halves), strides padded for bank-conflict-free
// b128 fragment loads (row stride bytes must be 16B-aligned and not 0 mod 256B).
#define W2S_STRIDE  520   // 256 rows  (1040 B/row)  == TDM pad: 16B per 1024B
#define SPK1_STRIDE 520   // 32 rows
#define SPK2_STRIDE 264   // 32 rows   (528 B/row)
#define WOS_STRIDE  264   // 2 rows
#define XS_STRIDE   40    // 32 rows   (80 B/row)

#define OFF_W2S    0
#define OFF_SPK1   (H2D * W2S_STRIDE)              // 133120
#define OFF_SPK2   (OFF_SPK1 + 32 * SPK1_STRIDE)   // 149760
#define OFF_XS     (OFF_SPK2 + 32 * SPK2_STRIDE)   // 158208
#define OFF_WOS    (OFF_XS + 32 * XS_STRIDE)       // 159488
#define LDS_HALVES (OFF_WOS + NCD * WOS_STRIDE)    // 160016
#define LDS_BYTES  (LDS_HALVES * 2)                // 320032 B  (< 320 KB WGP limit)

#if __has_builtin(__builtin_amdgcn_tensor_load_to_lds) && \
    __has_builtin(__builtin_amdgcn_s_wait_tensorcnt)
#define HAVE_TDM 1
#else
#define HAVE_TDM 0
#endif

__device__ __forceinline__ v16h ldh16(const _Float16* p0, const _Float16* p1) {
  v8h a = *(const v8h*)p0;
  v8h b = *(const v8h*)p1;
  return __builtin_shufflevector(a, b, 0,1,2,3,4,5,6,7,8,9,10,11,12,13,14,15);
}

__device__ __forceinline__ v8f wmma_f16(v16h a, v16h b, v8f c) {
  // D = A(16x32 f16) x B(32x16 f16) + C(16x16 f32)
  return __builtin_amdgcn_wmma_f32_16x16x32_f16(false, a, false, b, (short)0, c,
                                                false, false);
}

__device__ __forceinline__ v8f zero8f() {
  v8f v;
#pragma unroll
  for (int i = 0; i < 8; ++i) v[i] = 0.0f;
  return v;
}

// ---------------------------------------------------------------------------
// Kernel 0: f32 -> f16 weight conversion into workspace.
// ws layout (halves): W1h [512*1024] | W2h [256*512] | Woh [2*256]
// ---------------------------------------------------------------------------
__global__ void snn_convert_weights(const float* __restrict__ W1,
                                    const float* __restrict__ W2,
                                    const float* __restrict__ Wo,
                                    _Float16* __restrict__ ws) {
  const int N1 = H1D * FD;    // 524288
  const int N2 = H2D * H1D;   // 131072
  const int N3 = NCD * H2D;   // 512
  int i = blockIdx.x * blockDim.x + threadIdx.x;
  if (i < N1)                 ws[i] = (_Float16)W1[i];
  else if (i < N1 + N2)       ws[i] = (_Float16)W2[i - N1];
  else if (i < N1 + N2 + N3)  ws[i] = (_Float16)Wo[i - N1 - N2];
}

// ---------------------------------------------------------------------------
// Kernel 1: fully fused SNN. One block owns 32 batch rows for all 35 steps.
// Membranes live in WMMA C-fragment registers; spikes route through LDS.
// W2 is staged into LDS by the Tensor Data Mover (async, overlapped with
// the phase-1 GEMM), with TDM LDS-padding producing the 520-half stride.
// ---------------------------------------------------------------------------
__global__ void __launch_bounds__(256, 1)
snn_fused(const float* __restrict__ x,
          const _Float16* __restrict__ W1h,
          const float* __restrict__ b1,
          const _Float16* __restrict__ W2h,
          const float* __restrict__ b2,
          const _Float16* __restrict__ Woh,
          const float* __restrict__ bo,
          float* __restrict__ out) {
  extern __shared__ _Float16 smem[];
  _Float16* w2s  = smem + OFF_W2S;
  _Float16* spk1 = smem + OFF_SPK1;
  _Float16* spk2 = smem + OFF_SPK2;
  _Float16* xs   = smem + OFF_XS;
  _Float16* wos  = smem + OFF_WOS;

  const int tid     = threadIdx.x;
  const int w       = tid >> 5;        // wave id 0..7
  const int lane    = tid & 31;
  const int lm      = lane & 15;       // N within a 16-wide tile (C-layout)
  const int lhi     = lane >> 4;       // lane-group: selects K/M halves
  const int rowbase = blockIdx.x * 32;

  // ---- Phase 0: kick off async W2 staging (TDM), stage Wo manually.
#if HAVE_TDM
  if (w == 0) {
    // D# group 0: count=1 | lds_addr | global_addr[56:0] | type=2
    unsigned long long ga = (unsigned long long)(const void*)W2h;
    u32x4 g0;
    g0[0] = 1u;                                    // count=1, user mode
    g0[1] = (unsigned)(OFF_W2S * 2);               // LDS byte offset (=0)
    g0[2] = (unsigned)(ga & 0xFFFFFFFFu);
    g0[3] = (unsigned)((ga >> 32) & 0x1FFFFFFu) | (2u << 30);  // type=2
    // D# group 1: data_size=8B units; 2D tile 128x256 units, contiguous rows;
    // pad 4 DWORDs (16B) into LDS every 256 DWORDs (1024B) -> 1040B stride.
    i32x8 g1;
    g1[0] = (3 << 16)      // data_size = 8 bytes
          | (1 << 20)      // pad_enable
          | (7 << 22)      // pad_interval = 256 DWORDs
          | (3 << 25);     // pad_amount   = 4 DWORDs
    g1[1] = (128 << 16);   // tensor_dim0[15:0] = 128 units/row
    g1[2] = (256 << 16);   // tensor_dim0 hi=0 | tensor_dim1[15:0] = 256 rows
    g1[3] = (128 << 16);   // tensor_dim1 hi=0 | tile_dim0 = 128 units
    g1[4] = 256;           // tile_dim1 = 256 rows | tile_dim2 = 0
    g1[5] = 128;           // tensor_dim0_stride[31:0] = 128 units
    g1[6] = 0;             // stride0 hi | tensor_dim1_stride lo
    g1[7] = 0;
    i32x4 gz4;
    gz4[0] = 0; gz4[1] = 0; gz4[2] = 0; gz4[3] = 0;  // 2D: groups 2/3 unused
    i32x8 gz8;
#pragma unroll
    for (int i = 0; i < 8; ++i) gz8[i] = 0;
    __builtin_amdgcn_tensor_load_to_lds(g0, g1, gz4, gz4, gz8, 0);
  }
#else
  for (int v = tid; v < H2D * (H1D / 8); v += 256) {   // 16384 vec8 moves
    int r  = v >> 6;
    int c8 = (v & 63) * 8;
    *(v8h*)(w2s + r * W2S_STRIDE + c8) = *(const v8h*)(W2h + r * H1D + c8);
  }
#endif
  if (tid < 64) {
    int r  = tid >> 5;
    int c8 = (tid & 31) * 8;
    *(v8h*)(wos + r * WOS_STRIDE + c8) = *(const v8h*)(Woh + r * H2D + c8);
  }

  // ---- bias preloads (each lane needs bias only at its N = lm column)
  const int mt2 = w & 1;           // GEMM2/phase1 M-tile of this wave
  const int nb2 = (w >> 1) * 64;   // GEMM2 N-base of this wave (4 tiles)
  float b2r[4];
#pragma unroll
  for (int j = 0; j < 4; ++j) b2r[j] = b2[nb2 + j * 16 + lm];
  const float bor = (lm < NCD) ? bo[lm] : 0.0f;

  // ---- Phase 1: cur1 tile = x @ W1^T  (M=32 rows, N=512 cols, K=1024)
  // Runs while the TDM streams W2 into LDS in the background.
  // wave w: rows [16*(w&1), +16), cols [(w>>1)*128, +128) -> 8 N-tiles
  const int mh = w & 1;
  const int ng = w >> 1;
  v8f cur1[8];
#pragma unroll
  for (int nt = 0; nt < 8; ++nt) cur1[nt] = zero8f();

  __syncthreads();  // xs region protocol starts here

  for (int kc = 0; kc < FD / 32; ++kc) {
    const int k = kc * 32;
    {  // stage 32x32 f32 x-tile -> f16 LDS
      int r  = tid >> 3;
      int c4 = (tid & 7) * 4;
      const float* xp = x + (size_t)(rowbase + r) * FD + k + c4;
      float4 xv = *(const float4*)xp;
      if (kc + 1 < FD / 32) __builtin_prefetch(xp + 32, 0, 0);
      _Float16* d = xs + r * XS_STRIDE + c4;
      d[0] = (_Float16)xv.x; d[1] = (_Float16)xv.y;
      d[2] = (_Float16)xv.z; d[3] = (_Float16)xv.w;
    }
    __syncthreads();
    // A fragment (16-bit A layout: lanes<16 hold K {0..7,16..23}, lanes>=16 {8..15,24..31})
    const _Float16* ap = xs + (mh * 16 + lm) * XS_STRIDE + (lhi ? 8 : 0);
    v16h a = ldh16(ap, ap + 16);
#pragma unroll
    for (int nt = 0; nt < 8; ++nt) {
      int n = ng * 128 + nt * 16 + lm;
      // B[k][n] = W1[n][k]: same K split as A (lanes<16 K {0..7,16..23}, lanes>=16 {8..15,24..31})
      const _Float16* bp = W1h + (size_t)n * FD + k + (lhi ? 8 : 0);
      v16h bf = ldh16(bp, bp + 16);
      cur1[nt] = wmma_f16(a, bf, cur1[nt]);
    }
    __syncthreads();  // protect xs before next stage
  }
#pragma unroll
  for (int nt = 0; nt < 8; ++nt) {
    float bb = b1[ng * 128 + nt * 16 + lm];
#pragma unroll
    for (int i = 0; i < 8; ++i) cur1[nt][i] += bb;
  }

#if HAVE_TDM
  // Fence the async W2 tile before anyone reads w2s.
  if (w == 0) __builtin_amdgcn_s_wait_tensorcnt(0);
#endif
  __syncthreads();  // all waves see staged W2 (+Wo)

  // ---- persistent membrane state in fragment layout
  v8f m1[8];
#pragma unroll
  for (int nt = 0; nt < 8; ++nt) m1[nt] = zero8f();
  v8f m2[4];
#pragma unroll
  for (int j = 0; j < 4; ++j) m2[j] = zero8f();
  v8f mo = zero8f();

  const size_t memoff = (size_t)NUM_STEPS * BSZ * NCD;

  // ---- time-step loop: everything stays on-chip
  for (int s = 0; s < NUM_STEPS; ++s) {
    // layer-1 leaky (registers) -> spk1 LDS (f16)
#pragma unroll
    for (int nt = 0; nt < 8; ++nt) {
      int N = ng * 128 + nt * 16 + lm;
#pragma unroll
      for (int i = 0; i < 8; ++i) {
        float m   = m1[nt][i];
        float rst = (m > THRESH) ? THRESH : 0.0f;
        m = BETA * m + cur1[nt][i] - rst;
        m1[nt][i] = m;
        int M = mh * 16 + lhi * 8 + i;  // C-layout: lanes>=16 carry M+8
        spk1[M * SPK1_STRIDE + N] = (_Float16)((m > THRESH) ? 1.0f : 0.0f);
      }
    }
    __syncthreads();  // (A) spk1 ready; also fences last step's spk2 readers

    // layer-2 GEMM: [32x512] @ [512x256], accumulators seeded with b2
    v8f acc[4];
#pragma unroll
    for (int j = 0; j < 4; ++j)
#pragma unroll
      for (int i = 0; i < 8; ++i) acc[j][i] = b2r[j];

    for (int kc = 0; kc < H1D / 32; ++kc) {
      const int k = kc * 32;
      const _Float16* ap = spk1 + (mt2 * 16 + lm) * SPK1_STRIDE + k + (lhi ? 8 : 0);
      v16h a = ldh16(ap, ap + 16);
#pragma unroll
      for (int j = 0; j < 4; ++j) {
        const _Float16* bp = w2s + (nb2 + j * 16 + lm) * W2S_STRIDE + k + (lhi ? 8 : 0);
        v16h bf = ldh16(bp, bp + 16);
        acc[j] = wmma_f16(a, bf, acc[j]);
      }
    }
    // layer-2 leaky -> spk2 LDS
#pragma unroll
    for (int j = 0; j < 4; ++j) {
      int N = nb2 + j * 16 + lm;
#pragma unroll
      for (int i = 0; i < 8; ++i) {
        float m   = m2[j][i];
        float rst = (m > THRESH) ? THRESH : 0.0f;
        m = BETA * m + acc[j][i] - rst;
        m2[j][i] = m;
        int M = mt2 * 16 + lhi * 8 + i;
        spk2[M * SPK2_STRIDE + N] = (_Float16)((m > THRESH) ? 1.0f : 0.0f);
      }
    }
    __syncthreads();  // (B) spk2 ready

    // output GEMM: [32x256] @ [256x2] (Wo zero-padded to 16 cols), waves 0-1
    if (w < 2) {
      v8f acco;
#pragma unroll
      for (int i = 0; i < 8; ++i) acco[i] = bor;
#pragma unroll
      for (int kc = 0; kc < H2D / 32; ++kc) {
        const int k = kc * 32;
        const _Float16* ap = spk2 + (w * 16 + lm) * SPK2_STRIDE + k + (lhi ? 8 : 0);
        v16h a = ldh16(ap, ap + 16);
        v16h bf;
        if (lm < NCD) {
          const _Float16* bp = wos + lm * WOS_STRIDE + k + (lhi ? 8 : 0);
          bf = ldh16(bp, bp + 16);
        } else {
#pragma unroll
          for (int i = 0; i < 16; ++i) bf[i] = (_Float16)0.0f;
        }
        acco = wmma_f16(a, bf, acco);
      }
      // output leaky + store spk_rec / mem_rec
#pragma unroll
      for (int i = 0; i < 8; ++i) {
        float m   = mo[i];
        float rst = (m > THRESH) ? THRESH : 0.0f;
        m = BETA * m + acco[i] - rst;
        mo[i] = m;
        if (lm < NCD) {
          int row  = rowbase + w * 16 + lhi * 8 + i;
          size_t o = ((size_t)s * BSZ + row) * NCD + lm;
          out[o]          = (m > THRESH) ? 1.0f : 0.0f;
          out[memoff + o] = m;
        }
      }
    }
    // no trailing barrier needed: barrier (A) of the next step fences spk2
    // before its next writers (GEMM3 readers must pass (A) first).
  }
}

// ---------------------------------------------------------------------------
extern "C" void kernel_launch(void* const* d_in, const int* in_sizes, int n_in,
                              void* d_out, int out_size, void* d_ws, size_t ws_size,
                              hipStream_t stream) {
  (void)in_sizes; (void)n_in; (void)out_size; (void)ws_size;
  const float* x  = (const float*)d_in[0];
  const float* W1 = (const float*)d_in[1];
  const float* b1 = (const float*)d_in[2];
  const float* W2 = (const float*)d_in[3];
  const float* b2 = (const float*)d_in[4];
  const float* Wo = (const float*)d_in[5];
  const float* bo = (const float*)d_in[6];
  float* out = (float*)d_out;

  _Float16* W1h = (_Float16*)d_ws;          // 524288 halves
  _Float16* W2h = W1h + H1D * FD;           // 131072 halves
  _Float16* Woh = W2h + H2D * H1D;          // 512 halves   (~1.25 MB total ws)

  const int total = H1D * FD + H2D * H1D + NCD * H2D;
  snn_convert_weights<<<(total + 255) / 256, 256, 0, stream>>>(W1, W2, Wo, W1h);
  snn_fused<<<BSZ / 32, 256, LDS_BYTES, stream>>>(x, W1h, b1, W2h, b2, Woh, bo, out);
}
